// LlamaAttention_91336774517628
// MI455X (gfx1250) — matrix operand order repaired and checked by reading, not compiled, not run on hardware
//
#include <hip/hip_runtime.h>

#define B_ 2
#define S_ 2048
#define HIDDEN_ 4096
#define NH_ 32
#define NKV_ 8
#define HD_ 128
#define QKV_W 6144          // (32 + 2*8) * 128
#define KOFF 4096
#define VOFF 5120

typedef __attribute__((ext_vector_type(16))) __bf16 v16bf;
typedef __attribute__((ext_vector_type(2)))  __bf16 v2bf;
typedef __attribute__((ext_vector_type(8)))  float  v8f;
typedef __attribute__((ext_vector_type(4)))  unsigned int u32x4;
typedef __attribute__((ext_vector_type(4)))  int i32x4;

union FragU { u32x4 u[2]; v16bf v; };

#if __has_builtin(__builtin_amdgcn_global_load_async_to_lds_b128) && \
    __has_builtin(__builtin_amdgcn_s_wait_asynccnt)
#define HAS_ASYNC_LDS 1
#else
#define HAS_ASYNC_LDS 0
#endif

static __device__ __forceinline__ unsigned short f32_bf16(float f) {
  unsigned int u = __float_as_uint(f);
  u += 0x7fffu + ((u >> 16) & 1u);       // round-to-nearest-even
  return (unsigned short)(u >> 16);
}

#if __has_builtin(__builtin_amdgcn_cvt_pk_bf16_f32)
static __device__ __forceinline__ unsigned int pack2(float lo, float hi) {
  v2bf r = __builtin_amdgcn_cvt_pk_bf16_f32(lo, hi);
  return __builtin_bit_cast(unsigned int, r);
}
#else
static __device__ __forceinline__ unsigned int pack2(float lo, float hi) {
  return (unsigned int)f32_bf16(lo) | ((unsigned int)f32_bf16(hi) << 16);
}
#endif

// ---------------------------------------------------------------------------
// One-shot f32 -> bf16 conversion (RNE), 8 elements / thread, 128b stores.
// ---------------------------------------------------------------------------
__global__ __launch_bounds__(256) void convert_bf16(
    const float* __restrict__ src, unsigned short* __restrict__ dst, long long n)
{
  const long long i = ((long long)blockIdx.x * 256 + threadIdx.x) * 8;
  if (i >= n) return;
  float4 f0 = ((const float4*)(src + i))[0];
  float4 f1 = ((const float4*)(src + i))[1];
  u32x4 p;
  p[0] = pack2(f0.x, f0.y); p[1] = pack2(f0.z, f0.w);
  p[2] = pack2(f1.x, f1.y); p[3] = pack2(f1.z, f1.w);
  *(u32x4*)(dst + i) = p;
}

// ---------------------------------------------------------------------------
// GEMM: C[M,N] (f32) = A[M,K] (bf16, row-major) x B[N,K]^T (bf16, row-major)
// Double-buffered LDS pipeline. If available, staging uses CDNA5 async
// global->LDS (ASYNCcnt); else register-staged software pipelining.
// Block: 256 threads (8 waves). Tile: 128x256, BK=32. Wave: 64x64 (4x4 WMMA).
// Requires M%128==0, N%256==0, K%32==0.
// ---------------------------------------------------------------------------
#define BM 128
#define BN 256
#define BK 32
#define LDT 40   // padded LDS row stride (elements); 80 bytes, 16B-aligned rows

__global__ __launch_bounds__(256) void gemm_bf16_wmma(
    const unsigned short* __restrict__ A, const unsigned short* __restrict__ Bw,
    float* __restrict__ C, int M, int N, int K)
{
  __shared__ __attribute__((aligned(16))) unsigned short As[2][BM * LDT];
  __shared__ __attribute__((aligned(16))) unsigned short Bs[2][BN * LDT];

  const int t    = threadIdx.x;
  const int lane = t & 31;
  const int wave = t >> 5;
  const int wm   = (wave & 1) * 64;   // wave offset in M (2 groups)
  const int wn   = (wave >> 1) * 64;  // wave offset in N (4 groups)
  const int tileM = blockIdx.y * BM;
  const int tileN = blockIdx.x * BN;
  const int lm = lane & 15;
  const int hi = lane >> 4;

  const v8f zf = {0.f,0.f,0.f,0.f,0.f,0.f,0.f,0.f};
  v8f acc[4][4];
#pragma unroll
  for (int mi = 0; mi < 4; ++mi)
#pragma unroll
    for (int ni = 0; ni < 4; ++ni) acc[mi][ni] = zf;

  const int ra  = t >> 1;         // A row this thread stages
  const int ca  = (t & 1) * 16;   // A col offset (16 bf16 = 32B)
  const int nK  = K / BK;

#if HAS_ASYNC_LDS
  // ---- async staging: global -> LDS directly, tracked by ASYNCcnt ----
  auto stage_async = [&](int kt, int buf) {
    const unsigned short* sa = A + (size_t)(tileM + ra) * K + kt * BK + ca;
    unsigned short* da = &As[buf][ra * LDT + ca];
    __builtin_amdgcn_global_load_async_to_lds_b128(
        (i32x4*)(sa), (i32x4*)(da), 0, 0);
    __builtin_amdgcn_global_load_async_to_lds_b128(
        (i32x4*)(sa + 8), (i32x4*)(da + 8), 0, 0);
    const unsigned short* sb = Bw + (size_t)(tileN + t) * K + kt * BK;
    unsigned short* db = &Bs[buf][t * LDT];
    __builtin_amdgcn_global_load_async_to_lds_b128(
        (i32x4*)(sb), (i32x4*)(db), 0, 0);
    __builtin_amdgcn_global_load_async_to_lds_b128(
        (i32x4*)(sb + 8), (i32x4*)(db + 8), 0, 0);
    __builtin_amdgcn_global_load_async_to_lds_b128(
        (i32x4*)(sb + 16), (i32x4*)(db + 16), 0, 0);
    __builtin_amdgcn_global_load_async_to_lds_b128(
        (i32x4*)(sb + 24), (i32x4*)(db + 24), 0, 0);
  };
  stage_async(0, 0);
  __builtin_amdgcn_s_wait_asynccnt(0);
  __syncthreads();
#else
  // ---- register-staged software pipeline ----
  u32x4 aR0, aR1, bR0, bR1, bR2, bR3;
  auto load_regs = [&](int kt) {
    const unsigned short* sa = A + (size_t)(tileM + ra) * K + kt * BK + ca;
    aR0 = ((const u32x4*)sa)[0];
    aR1 = ((const u32x4*)sa)[1];
    const unsigned short* sb = Bw + (size_t)(tileN + t) * K + kt * BK;
    bR0 = ((const u32x4*)sb)[0];
    bR1 = ((const u32x4*)sb)[1];
    bR2 = ((const u32x4*)sb)[2];
    bR3 = ((const u32x4*)sb)[3];
  };
  auto store_lds = [&](int buf) {
    unsigned short* da = &As[buf][ra * LDT + ca];
    *(u32x4*)(da)     = aR0;
    *(u32x4*)(da + 8) = aR1;
    unsigned short* db = &Bs[buf][t * LDT];
    *(u32x4*)(db)      = bR0;
    *(u32x4*)(db + 8)  = bR1;
    *(u32x4*)(db + 16) = bR2;
    *(u32x4*)(db + 24) = bR3;
  };
  load_regs(0);
  store_lds(0);
  __syncthreads();
#endif

  for (int kt = 0; kt < nK; ++kt) {
    const int cur  = kt & 1;
    const bool more = (kt + 1 < nK);

#if HAS_ASYNC_LDS
    if (more) stage_async(kt + 1, cur ^ 1);   // overlap with WMMAs below
#else
    if (more) load_regs(kt + 1);              // issue globals before compute
    if (kt + 2 < nK)                          // warm L2 two tiles ahead
      __builtin_prefetch(A + (size_t)(tileM + ra) * K + (kt + 2) * BK + ca, 0, 0);
#endif

    // ---- fragments + WMMA (4x4) on buffer `cur` ----
    v16bf aF[4];
#pragma unroll
    for (int mi = 0; mi < 4; ++mi) {
      const unsigned short* p = &As[cur][(wm + mi * 16 + lm) * LDT + hi * 8];
      FragU u; u.u[0] = *(const u32x4*)p; u.u[1] = *(const u32x4*)(p + 16);
      aF[mi] = u.v;
    }
    v16bf bF[4];
#pragma unroll
    for (int ni = 0; ni < 4; ++ni) {
      const unsigned short* p = &Bs[cur][(wn + ni * 16 + lm) * LDT + hi * 8];
      FragU u; u.u[0] = *(const u32x4*)p; u.u[1] = *(const u32x4*)(p + 16);
      bF[ni] = u.v;
    }
#pragma unroll
    for (int mi = 0; mi < 4; ++mi)
#pragma unroll
      for (int ni = 0; ni < 4; ++ni)
        acc[mi][ni] = __builtin_amdgcn_wmma_f32_16x16x32_bf16(
            false, aF[mi], false, bF[ni], (short)0, acc[mi][ni], false, false);

    if (more) {
#if HAS_ASYNC_LDS
      __builtin_amdgcn_s_wait_asynccnt(0);
      __syncthreads();
#else
      __syncthreads();                        // everyone done reading buf nxt
      store_lds(cur ^ 1);
      __syncthreads();                        // stores visible for next iter
#endif
    }
  }

  // ---- store C (f32) ----
#pragma unroll
  for (int mi = 0; mi < 4; ++mi)
#pragma unroll
    for (int ni = 0; ni < 4; ++ni) {
      const int row0 = tileM + wm + mi * 16 + hi * 8;
      const int col  = tileN + wn + ni * 16 + lm;
#pragma unroll
      for (int rr = 0; rr < 8; ++rr)
        C[(size_t)(row0 + rr) * N + col] = acc[mi][ni][rr];
    }
}

// ---------------------------------------------------------------------------
// NeoX RoPE on q/k + f32 -> bf16 convert of the whole qkv buffer.
// ---------------------------------------------------------------------------
__global__ __launch_bounds__(256) void rope_convert(
    const float* __restrict__ qkv_f, const int* __restrict__ pos,
    unsigned short* __restrict__ qkv_b)
{
  const long long idx = (long long)blockIdx.x * blockDim.x + threadIdx.x;
  const long long total = (long long)B_ * S_ * QKV_W;
  if (idx >= total) return;
  const int o = (int)(idx % QKV_W);
  const long long bs = idx / QKV_W;
  const int s = (int)(bs % S_);
  const int b = (int)(bs / S_);
  const float x = qkv_f[idx];
  float val;
  if (o < VOFF) {                       // q [0,4096) or k [4096,5120): apply rope
    const int d = o & (HD_ - 1);
    const int p = pos[b * S_ + s];
    const int j = (d < 64) ? d : d - 64;
    const float inv_freq = __powf(10000.0f, -(float)j * (1.0f / 64.0f));
    const float ang = (float)p * inv_freq;
    float c, sn;
    __sincosf(ang, &sn, &c);            // (sin, cos)
    if (d < 64) {
      const float x2 = qkv_f[idx + 64];
      val = x * c - x2 * sn;
    } else {
      const float x1 = qkv_f[idx - 64];
      val = x * c + x1 * sn;
    }
  } else {
    val = x;                            // v: plain convert
  }
  qkv_b[idx] = f32_bf16(val);
}

// ---------------------------------------------------------------------------
// Flash attention (causal, GQA 4:1). Block = (query block of 128 rows, head, batch).
// 8 waves; wave w owns query rows [w*16, w*16+16). Key tiles of 64.
// Scores and P*V both via WMMA bf16. Online softmax with wave32 shfl reductions.
// Output written directly as bf16 (feeds GEMM2's A operand).
// ---------------------------------------------------------------------------
#define KT 64
#define KLD 136     // Ks row stride (elems)
#define VLD 72      // Vt / Ps row stride (elems)

__global__ __launch_bounds__(256) void attn_wmma(
    const unsigned short* __restrict__ qkv,   // bf16 (B*S, 6144)
    unsigned short* __restrict__ attnb)       // bf16 (B*S, 4096)
{
  __shared__ __attribute__((aligned(16))) unsigned short Ks[KT * KLD];     // 64 x 128 (+pad)
  __shared__ __attribute__((aligned(16))) unsigned short Vt[HD_ * VLD];    // 128 x 64 (+pad), V^T
  __shared__ __attribute__((aligned(16))) unsigned short Ps[8 * 16 * VLD]; // per-wave probs

  const int t    = threadIdx.x;
  const int lane = t & 31;
  const int wave = t >> 5;
  const int lm = lane & 15;
  const int hi = lane >> 4;
  const int qb  = blockIdx.x;
  const int h   = blockIdx.y;
  const int b   = blockIdx.z;
  const int hkv = h >> 2;                 // 32 q-heads -> 8 kv-heads
  const int qBase = qb * 128;

  // Q fragments (head_dim 128 = 4 chunks of K=32), kept in registers
  v16bf qF[4];
  {
    const int srow = qBase + wave * 16 + lm;
    const unsigned short* qp = qkv + (size_t)(b * S_ + srow) * QKV_W + h * HD_;
#pragma unroll
    for (int kc = 0; kc < 4; ++kc) {
      FragU u;
      u.u[0] = *(const u32x4*)(qp + kc * 32 + hi * 8);
      u.u[1] = *(const u32x4*)(qp + kc * 32 + hi * 8 + 16);
      qF[kc] = u.v;
    }
  }

  const v8f zf = {0.f,0.f,0.f,0.f,0.f,0.f,0.f,0.f};
  v8f O[8];
#pragma unroll
  for (int ni = 0; ni < 8; ++ni) O[ni] = zf;
  float ms[8], ls[8];
#pragma unroll
  for (int rr = 0; rr < 8; ++rr) { ms[rr] = -1.0e30f; ls[rr] = 0.f; }

  const int nTiles = (qBase + 128) / KT;  // causal: keys up to end of q block
  for (int kt = 0; kt < nTiles; ++kt) {
    const int kBase = kt * KT;
    __syncthreads();
    // ---- stage K tile (64 x 128) ----
    {
      const int kr = t >> 2, d0 = (t & 3) * 32;
      const unsigned short* src =
          qkv + (size_t)(b * S_ + kBase + kr) * QKV_W + KOFF + hkv * HD_ + d0;
      u32x4 x0 = ((const u32x4*)src)[0];
      u32x4 x1 = ((const u32x4*)src)[1];
      u32x4 x2 = ((const u32x4*)src)[2];
      u32x4 x3 = ((const u32x4*)src)[3];
      unsigned short* dst = &Ks[kr * KLD + d0];
      ((u32x4*)dst)[0] = x0; ((u32x4*)dst)[1] = x1;
      ((u32x4*)dst)[2] = x2; ((u32x4*)dst)[3] = x3;
    }
    // ---- stage V tile transposed (d-major: Vt[d][key]) ----
    {
      const int kr = t & 63, d1 = (t >> 6) * 32;
      const unsigned short* vs =
          qkv + (size_t)(b * S_ + kBase + kr) * QKV_W + VOFF + hkv * HD_ + d1;
      u32x4 y[4];
      y[0] = ((const u32x4*)vs)[0]; y[1] = ((const u32x4*)vs)[1];
      y[2] = ((const u32x4*)vs)[2]; y[3] = ((const u32x4*)vs)[3];
      const unsigned short* ye = (const unsigned short*)y;
#pragma unroll
      for (int i = 0; i < 32; ++i) Vt[(d1 + i) * VLD + kr] = ye[i];
    }
    __syncthreads();

    // ---- scores: S = Q K^T  (16 x 64 per wave) ----
    v8f sc[4];
#pragma unroll
    for (int ni = 0; ni < 4; ++ni) sc[ni] = zf;
#pragma unroll
    for (int kc = 0; kc < 4; ++kc) {
#pragma unroll
      for (int ni = 0; ni < 4; ++ni) {
        const unsigned short* p = &Ks[(ni * 16 + lm) * KLD + kc * 32 + hi * 8];
        FragU u; u.u[0] = *(const u32x4*)p; u.u[1] = *(const u32x4*)(p + 16);
        sc[ni] = __builtin_amdgcn_wmma_f32_16x16x32_bf16(
            false, qF[kc], false, u.v, (short)0, sc[ni], false, false);
      }
    }

    // ---- causal mask + online softmax ----
    const float scale = 0.08838834764831845f;   // 128^-0.5
    const int iRow0 = qBase + wave * 16 + hi * 8;
    float pv[4][8], rmax[8];
#pragma unroll
    for (int rr = 0; rr < 8; ++rr) rmax[rr] = -3.0e38f;
#pragma unroll
    for (int ni = 0; ni < 4; ++ni) {
      const int j = kBase + ni * 16 + lm;
#pragma unroll
      for (int rr = 0; rr < 8; ++rr) {
        const float v = (j <= iRow0 + rr) ? sc[ni][rr] * scale : -1.0e30f;
        pv[ni][rr] = v;
        rmax[rr] = fmaxf(rmax[rr], v);
      }
    }
#pragma unroll
    for (int m = 1; m < 16; m <<= 1)
#pragma unroll
      for (int rr = 0; rr < 8; ++rr)
        rmax[rr] = fmaxf(rmax[rr], __shfl_xor(rmax[rr], m, 32));
    float alpha[8], rsum[8];
#pragma unroll
    for (int rr = 0; rr < 8; ++rr) {
      const float nm = fmaxf(ms[rr], rmax[rr]);
      alpha[rr] = __expf(ms[rr] - nm);
      ms[rr] = nm;
      rsum[rr] = 0.f;
    }
#pragma unroll
    for (int ni = 0; ni < 4; ++ni)
#pragma unroll
      for (int rr = 0; rr < 8; ++rr) {
        const float p = __expf(pv[ni][rr] - ms[rr]);
        pv[ni][rr] = p;
        rsum[rr] += p;
      }
#pragma unroll
    for (int m = 1; m < 16; m <<= 1)
#pragma unroll
      for (int rr = 0; rr < 8; ++rr) rsum[rr] += __shfl_xor(rsum[rr], m, 32);
#pragma unroll
    for (int rr = 0; rr < 8; ++rr) ls[rr] = ls[rr] * alpha[rr] + rsum[rr];
#pragma unroll
    for (int ni = 0; ni < 8; ++ni)
#pragma unroll
      for (int rr = 0; rr < 8; ++rr) O[ni][rr] *= alpha[rr];

    // ---- write P (bf16) to per-wave LDS ----
    unsigned short* pw = &Ps[wave * 16 * VLD];
#pragma unroll
    for (int ni = 0; ni < 4; ++ni)
#pragma unroll
      for (int rr = 0; rr < 8; ++rr)
        pw[(rr + hi * 8) * VLD + ni * 16 + lm] = f32_bf16(pv[ni][rr]);

    // ---- O += P V  (K = 64 keys = 2 chunks of 32) ----
#pragma unroll
    for (int kc2 = 0; kc2 < 2; ++kc2) {
      FragU ua;
      const unsigned short* pp = &Ps[wave * 16 * VLD + lm * VLD + kc2 * 32 + hi * 8];
      ua.u[0] = *(const u32x4*)pp; ua.u[1] = *(const u32x4*)(pp + 16);
#pragma unroll
      for (int ni = 0; ni < 8; ++ni) {
        const unsigned short* vp = &Vt[(ni * 16 + lm) * VLD + kc2 * 32 + hi * 8];
        FragU ub; ub.u[0] = *(const u32x4*)vp; ub.u[1] = *(const u32x4*)(vp + 16);
        O[ni] = __builtin_amdgcn_wmma_f32_16x16x32_bf16(
            false, ua.v, false, ub.v, (short)0, O[ni], false, false);
      }
    }
  }

  // ---- epilogue: O / l -> attn (bf16, layout (b,s, h*128+d)) ----
  float inv[8];
#pragma unroll
  for (int rr = 0; rr < 8; ++rr) inv[rr] = 1.0f / ls[rr];
#pragma unroll
  for (int ni = 0; ni < 8; ++ni) {
    const int d = ni * 16 + lm;
#pragma unroll
    for (int rr = 0; rr < 8; ++rr) {
      const int i = qBase + wave * 16 + hi * 8 + rr;
      attnb[(size_t)(b * S_ + i) * (NH_ * HD_) + h * HD_ + d] =
          f32_bf16(O[ni][rr] * inv[rr]);
    }
  }
}

// ---------------------------------------------------------------------------
extern "C" void kernel_launch(void* const* d_in, const int* in_sizes, int n_in,
                              void* d_out, int out_size, void* d_ws, size_t ws_size,
                              hipStream_t stream) {
  (void)in_sizes; (void)n_in; (void)out_size; (void)ws_size;
  const int*   positions = (const int*)d_in[0];
  const float* hidden    = (const float*)d_in[1];
  const float* w_qkv     = (const float*)d_in[2];
  const float* w_o       = (const float*)d_in[3];
  float* out = (float*)d_out;

  // Workspace layout with aliasing (launches are stream-serialized):
  //   region A (48 MB) : w_qkv bf16  --later-->  w_o bf16
  //   region B (33.5MB): hidden bf16 --later-->  attn bf16
  //   region C (100.7MB): qkv f32
  //   region D (50.3MB): qkv bf16 (post-rope)
  char* ws = (char*)d_ws;
  const size_t BS = (size_t)B_ * S_;                         // 4096 rows
  unsigned short* regA  = (unsigned short*)ws;                              // 6144*4096 bf16
  unsigned short* regB  = (unsigned short*)(ws + (size_t)QKV_W * HIDDEN_ * 2);
  float*          qkv_f = (float*)(ws + (size_t)QKV_W * HIDDEN_ * 2
                                      + BS * HIDDEN_ * 2);
  unsigned short* qkv_b = (unsigned short*)((char*)qkv_f + BS * QKV_W * 4);

  dim3 blk(256);
  const long long nHid  = (long long)BS * HIDDEN_;        // 16.8M
  const long long nWqkv = (long long)QKV_W * HIDDEN_;     // 25.2M
  const long long nWo   = (long long)HIDDEN_ * HIDDEN_;   // 16.8M

  // 0) one-shot bf16 conversions
  convert_bf16<<<(int)(nHid  / 8 / 256), blk, 0, stream>>>(hidden, regB, nHid);
  convert_bf16<<<(int)(nWqkv / 8 / 256), blk, 0, stream>>>(w_qkv, regA, nWqkv);
  // 1) qkv = hidden x w_qkv^T   (M=4096, N=6144, K=4096)
  gemm_bf16_wmma<<<dim3(QKV_W / BN, (int)BS / BM), blk, 0, stream>>>(
      regB, regA, qkv_f, (int)BS, QKV_W, HIDDEN_);
  // 2) rope + bf16 convert
  const long long total = (long long)BS * QKV_W;
  rope_convert<<<(int)((total + 255) / 256), blk, 0, stream>>>(qkv_f, positions, qkv_b);
  // 2b) w_o bf16 (reuses region A; GEMM1 already consumed w_qkv bf16)
  convert_bf16<<<(int)(nWo / 8 / 256), blk, 0, stream>>>(w_o, regA, nWo);
  // 3) flash attention -> bf16 attn (reuses region B)
  attn_wmma<<<dim3(S_ / 128, NH_, B_), blk, 0, stream>>>(qkv_b, regB);
  // 4) out = attn x w_o^T       (M=4096, N=4096, K=4096)
  gemm_bf16_wmma<<<dim3(HIDDEN_ / BN, (int)BS / BM), blk, 0, stream>>>(
      regB, regA, out, (int)BS, HIDDEN_, HIDDEN_);
}
